// loss_fn_40948218200546
// MI455X (gfx1250) — matrix operand order repaired and checked by reading, not compiled, not run on hardware
//
#include <hip/hip_runtime.h>
#include <hip/hip_bf16.h>
#include <math.h>

// InfoNCE fused GEMM + online logsumexp for MI455X (gfx1250, wave32, WMMA).
//
// neg_dist = (ref @ neg^T) * invT is 16384x16384x128 = 68.7 GFLOP, never
// materialized (would be ~2 GB of HBM traffic vs 25 MB of inputs). fp32
// inputs are split into bf16 hi/lo pairs (x ~= hi + lo); each K-block uses
// 3 bf16 WMMAs (hi*hi + hi*lo + lo*hi) accumulating in f32 -> ~2^-16
// relative error at ~8x the fp32 matrix-op rate: 12 v_wmma_f32_16x16x32_bf16
// per 16x16x128 tile (vs 32 v_wmma_f32_16x16x4_f32).
//
// Grid: 256 blocks x 4 waves (BM=64 rows/block, 16-row strip per wave) so a
// ~256-WGP part is covered; 104.4 KB LDS/block -> 3 blocks (12 waves)/WGP.
// Each block streams neg_z (8 MB, L2-resident in 192 MB) through LDS in
// 128-column chunks. Online logsumexp state is PER-LANE (each lane owns its
// private column subset) so the hot loop has no cross-lane shuffles; the 16
// (m,s) pairs per row merge associatively once at the end. exp/log run in
// base 2 (log2e folded into inv_temp) to hit v_exp_f32/v_log_f32 directly.
//
// Staging goes through VGPRs (not async/TDM) on purpose: the LDS format is a
// transformed representation (fp32 -> split bf16 hi/lo), so VALU must touch
// the data anyway; the conversion hides under the WMMA chain.

typedef __attribute__((ext_vector_type(16))) __bf16 v16bf;
typedef __attribute__((ext_vector_type(8)))  float  v8f;

#define NPTS 16384
#define DDIM 128
#define LDB  136      // bf16 elems per LDS row (128 + 8 pad): 272B stride, 16B aligned
#define BM   64       // rows per block (4 waves x 16-row strips)
#define BN   128      // neg columns per chunk
#define THREADS 128

union FragU { struct { uint4 a, b; } u; v16bf v; };
__device__ inline v16bf make_frag(uint4 x, uint4 y) {
    FragU f; f.u.a = x; f.u.b = y; return f.v;
}

// Split 4 fp32 into packed bf16 hi (truncated) and bf16 lo (exact remainder
// truncated to bf16), store 8B each into the hi/lo LDS tiles.
__device__ inline void split_store4(float4 v, __bf16* hiP, __bf16* loP) {
    unsigned u0 = __float_as_uint(v.x), u1 = __float_as_uint(v.y);
    unsigned u2 = __float_as_uint(v.z), u3 = __float_as_uint(v.w);
    float h0 = __uint_as_float(u0 & 0xffff0000u);
    float h1 = __uint_as_float(u1 & 0xffff0000u);
    float h2 = __uint_as_float(u2 & 0xffff0000u);
    float h3 = __uint_as_float(u3 & 0xffff0000u);
    unsigned l0 = __float_as_uint(v.x - h0) >> 16;
    unsigned l1 = __float_as_uint(v.y - h1) >> 16;
    unsigned l2 = __float_as_uint(v.z - h2) >> 16;
    unsigned l3 = __float_as_uint(v.w - h3) >> 16;
    uint2 hi = make_uint2((u0 >> 16) | (u1 & 0xffff0000u),
                          (u2 >> 16) | (u3 & 0xffff0000u));
    uint2 lo = make_uint2(l0 | (l1 << 16), l2 | (l3 << 16));
    *(uint2*)hiP = hi;
    *(uint2*)loP = lo;
}

__global__ void zero_ws_kernel(float* __restrict__ ws) {
    if (threadIdx.x < 2) ws[threadIdx.x] = 0.0f;
}

// align = -(invT/N) * sum(ref_z * pos_z) -> flattened dot product (fp32 exact).
__global__ __launch_bounds__(256) void pos_dot_kernel(const float* __restrict__ a,
                                                      const float* __restrict__ b,
                                                      float* __restrict__ ws) {
    size_t i = ((size_t)blockIdx.x * 256u + threadIdx.x) * 4u;
    float4 va = *(const float4*)(a + i);
    float4 vb = *(const float4*)(b + i);
    float acc = va.x * vb.x + va.y * vb.y + va.z * vb.z + va.w * vb.w;
#pragma unroll
    for (int off = 1; off < 32; off <<= 1)
        acc += __shfl_xor(acc, off, 32);
    if ((threadIdx.x & 31) == 0) atomicAdd(ws + 0, acc);
}

__global__ __launch_bounds__(THREADS) void infonce_lse_kernel(const float* __restrict__ refZ,
                                                              const float* __restrict__ negZ,
                                                              const float* __restrict__ logInvT,
                                                              float* __restrict__ ws) {
    extern __shared__ __bf16 lds[];
    __bf16* refHiS = lds;                    // BM x LDB
    __bf16* refLoS = refHiS + BM * LDB;
    __bf16* negHiS = refLoS + BM * LDB;      // BN x LDB
    __bf16* negLoS = negHiS + BN * LDB;

    const int tid  = threadIdx.x;
    const int lane = tid & 31;
    const int wave = tid >> 5;                  // 0..3 -> 16-row strip
    const int rowBase = blockIdx.x * BM;
    // base-2 online softmax: fold log2(e) into the temperature scale
    const float invT2 = __expf(logInvT[0]) * 1.44269504088896340736f;

    // Stage + split this block's 64 ref rows (coalesced b128 loads).
    for (int i = tid; i < BM * (DDIM / 4); i += THREADS) {
        int r  = i >> 5;       // 32 float4 per row
        int c4 = i & 31;
        float4 v = *(const float4*)(refZ + (size_t)(rowBase + r) * DDIM + c4 * 4);
        split_store4(v, refHiS + r * LDB + c4 * 4, refLoS + r * LDB + c4 * 4);
    }
    __syncthreads();

    // A-matrix (16x32 bf16) per ISA 7.12.2: lanes 0-15 hold M=lane with
    // K = {kb*32 + 0..7, kb*32 + 16..23}; lanes 16-31 the +8 K runs.
    // Hoist all 4 K-blocks x {hi,lo} = 64 VGPRs; A is fixed for the kernel.
    const int mrow = wave * 16 + (lane & 15);
    const int asel = (lane < 16) ? 0 : 8;
    v16bf aHi[4], aLo[4];
#pragma unroll
    for (int kb = 0; kb < 4; ++kb) {
        const __bf16* ph = refHiS + mrow * LDB + kb * 32 + asel;
        const __bf16* pl = refLoS + mrow * LDB + kb * 32 + asel;
        aHi[kb] = make_frag(*(const uint4*)ph, *(const uint4*)(ph + 16));
        aLo[kb] = make_frag(*(const uint4*)pl, *(const uint4*)(pl + 16));
    }

    // Per-lane online logsumexp state (C layout: VGPR r -> row r + (lane<16?0:8),
    // column = lane&15 within each 16-wide subtile). No shuffles in hot loop.
    float m[8], s[8];
#pragma unroll
    for (int r = 0; r < 8; ++r) { m[r] = -1.0e30f; s[r] = 0.0f; }

    for (int c0 = 0; c0 < NPTS; c0 += BN) {
        __syncthreads();  // previous chunk's compute done before overwrite
        for (int i = tid; i < BN * (DDIM / 4); i += THREADS) {
            int r  = i >> 5;
            int c4 = i & 31;
            float4 v = *(const float4*)(negZ + (size_t)(c0 + r) * DDIM + c4 * 4);
            split_store4(v, negHiS + r * LDB + c4 * 4, negLoS + r * LDB + c4 * 4);
        }
        __syncthreads();

#pragma unroll
        for (int st = 0; st < BN / 16; ++st) {
            // B-matrix (32x16 bf16): lane holds column N=(lane&15), K runs
            // {+0..7, +16..23} (+8 for lanes 16-31), matching A -> two ds_load_b128.
            const int bcol = st * 16 + (lane & 15);
            const int bsel = (lane < 16) ? 0 : 8;
            v8f c = {};
#pragma unroll
            for (int kb = 0; kb < 4; ++kb) {
                const __bf16* ph = negHiS + bcol * LDB + kb * 32 + bsel;
                const __bf16* pl = negLoS + bcol * LDB + kb * 32 + bsel;
                v16bf bHi = make_frag(*(const uint4*)ph, *(const uint4*)(ph + 16));
                v16bf bLo = make_frag(*(const uint4*)pl, *(const uint4*)(pl + 16));
                // x*y ~= hi*hi + hi*lo + lo*hi  (lo*lo ~ 2^-16, dropped)
                c = __builtin_amdgcn_wmma_f32_16x16x32_bf16(
                        false, aLo[kb], false, bHi, (short)0, c, false, false);
                c = __builtin_amdgcn_wmma_f32_16x16x32_bf16(
                        false, aHi[kb], false, bLo, (short)0, c, false, false);
                c = __builtin_amdgcn_wmma_f32_16x16x32_bf16(
                        false, aHi[kb], false, bHi, (short)0, c, false, false);
            }
            // Per-lane online update, base 2: max, 2x v_exp_f32, fma.
#pragma unroll
            for (int r = 0; r < 8; ++r) {
                float x  = c[r] * invT2;
                float mn = fmaxf(m[r], x);
                s[r] = s[r] * exp2f(m[r] - mn) + exp2f(x - mn);
                m[r] = mn;
            }
        }
    }

    // Merge the 16 per-lane (m,s) pairs per row (associative logsumexp merge),
    // then one lane per group accumulates sum_r lse(row) into ws[1].
    float acc = 0.0f;
#pragma unroll
    for (int r = 0; r < 8; ++r) {
        float mm = m[r], ss = s[r];
#pragma unroll
        for (int off = 1; off < 16; off <<= 1) {
            float mo = __shfl_xor(mm, off, 16);
            float so = __shfl_xor(ss, off, 16);
            float mn = fmaxf(mm, mo);
            ss = ss * exp2f(mm - mn) + so * exp2f(mo - mn);
            mm = mn;
        }
        acc += mm + __log2f(ss);
    }
    if ((lane & 15) == 0)
        atomicAdd(ws + 1, acc * 0.69314718055994530942f);  // back to natural log
}

// JAX flattens dicts in sorted-key order: {loss, neg_loss, pos_loss}.
__global__ void finalize_kernel(const float* __restrict__ logInvT,
                                const float* __restrict__ ws,
                                float* __restrict__ out) {
    if (threadIdx.x == 0) {
        float invT   = __expf(logInvT[0]);
        float align  = -invT * ws[0] / (float)NPTS;   // pos_loss
        float unifrm = ws[1] / (float)NPTS;           // neg_loss
        out[0] = align + unifrm;                      // loss
        out[1] = unifrm;                              // neg_loss
        out[2] = align;                               // pos_loss
    }
}

extern "C" void kernel_launch(void* const* d_in, const int* in_sizes, int n_in,
                              void* d_out, int out_size, void* d_ws, size_t ws_size,
                              hipStream_t stream) {
    const float* ref     = (const float*)d_in[0];
    const float* pos     = (const float*)d_in[1];
    const float* neg     = (const float*)d_in[2];
    const float* logInvT = (const float*)d_in[3];
    float* out = (float*)d_out;
    float* ws  = (float*)d_ws;

    zero_ws_kernel<<<1, 32, 0, stream>>>(ws);

    // Flattened ref·pos dot: 2,097,152 elems / (256 thr * 4 elems) = 2048 blocks.
    pos_dot_kernel<<<(NPTS * DDIM) / 1024, 256, 0, stream>>>(ref, pos, ws);

    // Fused GEMM + logsumexp: 256 row-blocks of 64 rows, 4 waves each.
    // LDS = 2*(BM+BN)*LDB bf16 = 104,448 B -> 3 blocks (12 waves) per WGP.
    size_t ldsBytes = (size_t)(2 * (BM + BN) * LDB) * sizeof(__bf16);
    infonce_lse_kernel<<<NPTS / BM, THREADS, ldsBytes, stream>>>(ref, neg, logInvT, ws);

    finalize_kernel<<<1, 32, 0, stream>>>(logInvT, ws, out);
}
